// DAT3_1082331758600
// MI455X (gfx1250) — compile-verified, matrix operand order checked by reading, not run on hardware
//
#include <hip/hip_runtime.h>
#include <hip/hip_bf16.h>

// ---------------------------------------------------------------------------
// MI455X (gfx1250) implementation.
//
// Roofline: after pushing the linear W2 through the edge aggregation
// (agg_i = (sum_j w_ij relu(pre_ij)) @ W2 + (sum_j w_ij) b2), total work is
// ~1.5 GFLOP on a working set of a few MB -> fully L2 resident (192 MB L2,
// 23.3 TB/s HBM never touched after the first pass). All dense 512xKx128
// GEMMs go through v_wmma_f32_16x16x32_bf16 (bf16 in, f32 accum); irregular
// parts (masked softmax, leaky-relu, LN, tanh joint-attention, InfoNCE) are
// fused wave32 VALU kernels.
//
// WMMA data movement (fixed after round-0 disasm showed scalarized
// ds_load_u16 gathers): A fragments are loaded straight from global as four
// global_load_b128 per lane (each wave owns its 16 rows exclusively); the
// shared B tile is staged into LDS in fragment-permuted order
// (perm(k) = ((k>>3)&1)*16 + (k>>4)*8 + (k&7)) so each lane's 16 bf16
// fragment elements are 32 contiguous bytes -> two ds_load_b128.
//
// Input flattening (jax pytree, dict keys sorted):
//  0 h_1d_seq (2,512,128)  1 drug_seq (2,64,128)  2 prot_node_feat (2,256,3)
//  3 prot_adj (2,256,256)  4 prot_dist (2,256,256)
//  params: 5..12  cl{b11,b12,b31,b32,w11,w12,w31,w32}
//          13..22 cross{bk,bo,bq,bv,ln_b,ln_g,wk,wo,wq,wv}
//          23 gnn.in_b 24 gnn.in_w
//          25+8l.. gnn.layers[l]{W1,W2,Wa,Wu,b1,b2,ba,bu}  (l=0..2)
//          49 gnn.out_b 50 gnn.out_w
//          51..55 joint{wa,wb,wd,wp,ws}
//          56..63 pred{b1,b2,b3,b4,w1,w2,w3,w4}
// Output: d_out[0..1] = y_pred, d_out[2] = cl_loss.
// ---------------------------------------------------------------------------

typedef __attribute__((ext_vector_type(16))) __bf16 v16bf;
typedef __attribute__((ext_vector_type(8)))  __bf16 v8bf;
typedef __attribute__((ext_vector_type(8)))  float  v8f;

// -------------------- block-wide reductions (blockDim.x == 128) ------------
__device__ __forceinline__ float breduce_sum(float* red, int t, float v) {
  red[t] = v; __syncthreads();
  for (int off = 64; off > 0; off >>= 1) {
    if (t < off) red[t] += red[t + off];
    __syncthreads();
  }
  float r = red[0]; __syncthreads();
  return r;
}
__device__ __forceinline__ float breduce_max(float* red, int t, float v) {
  red[t] = v; __syncthreads();
  for (int off = 64; off > 0; off >>= 1) {
    if (t < off) red[t] = fmaxf(red[t], red[t + off]);
    __syncthreads();
  }
  float r = red[0]; __syncthreads();
  return r;
}

// K-permutation that makes a lane's WMMA fragment contiguous in LDS:
// fragment elements for lane-half h are k in {h*8..h*8+7} U {16+h*8..16+h*8+7},
// which map to positions h*16 .. h*16+15.
__device__ __forceinline__ int kperm(int k) {
  return (((k >> 3) & 1) << 4) + ((k >> 4) << 3) + (k & 7);
}

// -------------------- generic WMMA GEMM ------------------------------------
// C[M,N] = act( A[M,K] @ W[K,N] + bias + addend ).  M,N multiples of 64,
// K multiple of 32, ldw == 128 in all uses here.  Block = 128 threads
// (4 wave32), tile 64x64; each wave computes a 16x64 strip as 4 16x16 f32
// accumulators via v_wmma_f32_16x16x32_bf16.
__global__ __launch_bounds__(128)
void gemm_wmma(const float* __restrict__ A, int lda,
               const float* __restrict__ W, int ldw,
               const float* __restrict__ bias,
               const float* __restrict__ addend, int ldadd,
               float* __restrict__ C, int ldc,
               int K, int act) {
  // B tile in fragment-permuted layout; row stride 40 bf16 (80 B) keeps the
  // two 16-byte fragment halves aligned and spreads LDS banks.
  __shared__ __bf16 Bs[64][40];
  const int t = threadIdx.x;
  const int wid = t >> 5;
  const int lane = t & 31;
  const int rowBase = blockIdx.x * 64;
  const int colBase = blockIdx.y * 64;

  v8f acc[4];
  acc[0] = {}; acc[1] = {}; acc[2] = {}; acc[3] = {};

  const int arow = rowBase + (wid << 4) + (lane & 15);  // this lane's A row
  const int khalf = (lane >> 4) << 3;                   // 0 or 8
  const int kfrag = (lane >> 4) << 4;                   // 0 or 16 (LDS frag base)

  for (int k0 = 0; k0 < K; k0 += 32) {
    // ---- stage 32x64 W tile into permuted LDS layout ----
    #pragma unroll
    for (int i = 0; i < 8; i++) {
      int idx = i * 128 + t;          // 1024 float2 pairs
      int r = idx >> 5;               // k row 0..31
      int c2 = (idx & 31) * 2;        // even column 0..62
      const float2 wv = *(const float2*)&W[(k0 + r) * ldw + colBase + c2];
      int pk = kperm(r);
      Bs[c2][pk]     = (__bf16)wv.x;
      Bs[c2 + 1][pk] = (__bf16)wv.y;
    }
    __syncthreads();

    // ---- A fragment straight from global: two runs of 8 consecutive f32 ----
    const float* ap = &A[arow * lda + k0 + khalf];
    const float4 x0 = *(const float4*)(ap);
    const float4 x1 = *(const float4*)(ap + 4);
    const float4 x2 = *(const float4*)(ap + 16);
    const float4 x3 = *(const float4*)(ap + 20);
    v16bf afrag;
    afrag[0]  = (__bf16)x0.x; afrag[1]  = (__bf16)x0.y;
    afrag[2]  = (__bf16)x0.z; afrag[3]  = (__bf16)x0.w;
    afrag[4]  = (__bf16)x1.x; afrag[5]  = (__bf16)x1.y;
    afrag[6]  = (__bf16)x1.z; afrag[7]  = (__bf16)x1.w;
    afrag[8]  = (__bf16)x2.x; afrag[9]  = (__bf16)x2.y;
    afrag[10] = (__bf16)x2.z; afrag[11] = (__bf16)x2.w;
    afrag[12] = (__bf16)x3.x; afrag[13] = (__bf16)x3.y;
    afrag[14] = (__bf16)x3.z; afrag[15] = (__bf16)x3.w;

    // ---- 4 WMMAs; each B fragment is two contiguous 16B LDS loads ----
    #pragma unroll
    for (int c4 = 0; c4 < 4; c4++) {
      const int ncol = (c4 << 4) + (lane & 15);
      const __bf16* bp = &Bs[ncol][kfrag];
      v8bf blo = *(const v8bf*)bp;
      v8bf bhi = *(const v8bf*)(bp + 8);
      v16bf bfrag;
      #pragma unroll
      for (int e = 0; e < 8; e++) { bfrag[e] = blo[e]; bfrag[e + 8] = bhi[e]; }
      acc[c4] = __builtin_amdgcn_wmma_f32_16x16x32_bf16(
          false, afrag, false, bfrag, (short)0, acc[c4], false, false);
    }
    __syncthreads();
  }

  // epilogue: VGPR r of C holds M = r + 8*(lane>=16), N = lane&15
  const int mbase = rowBase + (wid << 4) + ((lane >> 4) << 3);
  const int nloc = lane & 15;
  #pragma unroll
  for (int c4 = 0; c4 < 4; c4++) {
    int col = colBase + (c4 << 4) + nloc;
    float bv = bias ? bias[col] : 0.0f;
    #pragma unroll
    for (int r = 0; r < 8; r++) {
      int row = mbase + r;
      float v = acc[c4][r] + bv;
      if (addend) v += addend[row * ldadd + col];
      if (act == 1) v = fmaxf(v, 0.0f);
      C[row * ldc + col] = v;
    }
  }
}

// -------------------- GNN input projection (K=3) ---------------------------
__global__ void gnn_in_kernel(const float* __restrict__ nf,
                              const float* __restrict__ w,
                              const float* __restrict__ b,
                              float* __restrict__ hc) {
  int row = blockIdx.x, t = threadIdx.x;
  float v = nf[row * 3 + 0] * w[0 * 128 + t]
          + nf[row * 3 + 1] * w[1 * 128 + t]
          + nf[row * 3 + 2] * w[2 * 128 + t] + b[t];
  hc[row * 512 + t] = v;   // column block 0 of the dense-concat buffer
}

// -------------------- attention scalar projections s_i, s_j ----------------
__global__ void attn_proj_kernel(const float* __restrict__ hc,
                                 const float* __restrict__ Wa, int in_d,
                                 float* __restrict__ si, float* __restrict__ sj) {
  __shared__ float red[128];
  int row = blockIdx.x, t = threadIdx.x;
  float a1 = 0.f, a2 = 0.f;
  for (int k = t; k < in_d; k += 128) {
    float h = hc[row * 512 + k];
    a1 += h * Wa[k];
    a2 += h * Wa[in_d + k];
  }
  float r1 = breduce_sum(red, t, a1);
  float r2 = breduce_sum(red, t, a2);
  if (t == 0) { si[row] = r1; sj[row] = r2; }
}

// -------------------- fused edge kernel: masked softmax + weighted relu agg
__global__ void gnn_edge_kernel(const float* __restrict__ si,
                                const float* __restrict__ sj,
                                const float* __restrict__ dist,
                                const float* __restrict__ adj,
                                const float* __restrict__ a_i,
                                const float* __restrict__ a_j,
                                const float* __restrict__ w1d,
                                const float* __restrict__ b1,
                                const float* __restrict__ wadp,
                                const float* __restrict__ bap,
                                float* __restrict__ pagg,
                                float* __restrict__ sumw) {
  __shared__ float sL[256], wL[256], dL[256], red[128];
  int bi = blockIdx.x;          // b*256 + i
  int b = bi >> 8;
  int t = threadIdx.x;
  float wad = wadp[0], ba = bap[0], s_i = si[bi];
  float ad[2];
  #pragma unroll
  for (int u = 0; u < 2; u++) {
    int j = t + u * 128;
    float d = dist[bi * 256 + j];
    ad[u] = adj[bi * 256 + j];
    dL[j] = d;
    float s = s_i + sj[b * 256 + j] + d * wad + ba;
    s = (s > 0.f) ? s : 0.2f * s;                 // leaky_relu 0.2
    sL[j] = (ad[u] == 0.f) ? -1.0e9f : s;         // NEG_MASK
  }
  __syncthreads();
  float mx = breduce_max(red, t, fmaxf(sL[t], sL[t + 128]));
  float e0 = __expf(sL[t] - mx), e1 = __expf(sL[t + 128] - mx);
  float den = breduce_sum(red, t, e0 + e1);
  float inv = 1.0f / den;
  wL[t] = e0 * inv * ad[0];
  wL[t + 128] = e1 * inv * ad[1];
  float sw = breduce_sum(red, t, wL[t] + wL[t + 128]);  // also barriers wL
  if (t == 0) sumw[bi] = sw;
  // pass B: p_agg[bi,t] = sum_j w_j * relu(a_i + a_j + d*w1d + b1)
  float aid = a_i[bi * 128 + t], w1 = w1d[t], bb = b1[t];
  float acc = 0.f;
  for (int j = 0; j < 256; j++) {
    float pre = aid + a_j[(b * 256 + j) * 128 + t] + dL[j] * w1 + bb;
    acc += wL[j] * fmaxf(pre, 0.f);
  }
  pagg[bi * 128 + t] = acc;
}

// agg += sumw[row] * b2[col]   (linear-deferred bias of msg MLP)
__global__ void addsb_kernel(float* __restrict__ agg,
                             const float* __restrict__ sumw,
                             const float* __restrict__ b2) {
  int row = blockIdx.x, t = threadIdx.x;
  agg[row * 128 + t] += sumw[row] * b2[t];
}

// -------------------- cross attention (8 heads, hd=16) ---------------------
__global__ void mha_kernel(const float* __restrict__ q,
                           const float* __restrict__ k,
                           const float* __restrict__ v,
                           float* __restrict__ o) {
  __shared__ float sc[512], red[128], oacc[128][16], qv[16];
  int bid = blockIdx.x;                 // (b, h, n)
  int b = bid >> 11;
  int rem = bid & 2047;
  int h = rem >> 8;
  int n = rem & 255;
  int t = threadIdx.x;
  if (t < 16) qv[t] = q[(b * 256 + n) * 128 + h * 16 + t];
  __syncthreads();
  float lmax = -1e30f;
  #pragma unroll
  for (int u = 0; u < 4; u++) {
    int l = t + u * 128;
    const float* kp = &k[(b * 512 + l) * 128 + h * 16];
    float s = 0.f;
    #pragma unroll
    for (int d = 0; d < 16; d++) s += qv[d] * kp[d];
    s *= 0.25f;                          // 1/sqrt(16)
    sc[l] = s;
    lmax = fmaxf(lmax, s);
  }
  __syncthreads();
  float mx = breduce_max(red, t, lmax);
  float lsum = 0.f;
  #pragma unroll
  for (int u = 0; u < 4; u++) {
    int l = t + u * 128;
    float e = __expf(sc[l] - mx);
    sc[l] = e;
    lsum += e;
  }
  __syncthreads();
  float den = breduce_sum(red, t, lsum);
  float inv = 1.0f / den;
  float ol[16];
  #pragma unroll
  for (int d = 0; d < 16; d++) ol[d] = 0.f;
  #pragma unroll
  for (int u = 0; u < 4; u++) {
    int l = t + u * 128;
    float p = sc[l] * inv;
    const float* vp = &v[(b * 512 + l) * 128 + h * 16];
    #pragma unroll
    for (int d = 0; d < 16; d++) ol[d] += p * vp[d];
  }
  #pragma unroll
  for (int d = 0; d < 16; d++) oacc[t][d] = ol[d];
  __syncthreads();
  for (int off = 64; off > 0; off >>= 1) {
    if (t < off) {
      #pragma unroll
      for (int d = 0; d < 16; d++) oacc[t][d] += oacc[t + off][d];
    }
    __syncthreads();
  }
  if (t < 16) o[(b * 256 + n) * 128 + h * 16 + t] = oacc[0][t];
}

// residual + layernorm
__global__ void ln_kernel(const float* __restrict__ x0,
                          const float* __restrict__ resid,
                          const float* __restrict__ g,
                          const float* __restrict__ bl,
                          float* __restrict__ out) {
  __shared__ float red[128];
  int row = blockIdx.x, t = threadIdx.x;
  float x = x0[row * 128 + t] + resid[row * 128 + t];
  float mean = breduce_sum(red, t, x) * (1.0f / 128.0f);
  float dv = x - mean;
  float var = breduce_sum(red, t, dv * dv) * (1.0f / 128.0f);
  out[row * 128 + t] = g[t] * dv * rsqrtf(var + 1e-5f) + bl[t];
}

// -------------------- joint attention --------------------------------------
__global__ void joint_kernel(const float* __restrict__ dproj,
                             const float* __restrict__ pproj,
                             const float* __restrict__ daM,
                             const float* __restrict__ pbM,
                             const float* __restrict__ jws,
                             float* __restrict__ xint) {
  __shared__ float dI[128], daI[128], wsL[128], A[256], red[128];
  int bid = blockIdx.x;                  // b*64 + i
  int b = bid >> 6;
  int t = threadIdx.x;
  dI[t]  = dproj[bid * 128 + t];
  daI[t] = daM[bid * 128 + t];
  wsL[t] = jws[t];
  __syncthreads();
  // scores: A[j] = sum_d tanh(dI + pproj[j]) * ws
  #pragma unroll
  for (int u = 0; u < 2; u++) {
    int j = t + u * 128;
    const float* pp = &pproj[(b * 256 + j) * 128];
    float acc = 0.f;
    for (int d = 0; d < 128; d++) acc += tanhf(dI[d] + pp[d]) * wsL[d];
    A[j] = acc;
  }
  __syncthreads();
  float mx = breduce_max(red, t, fmaxf(A[t], A[t + 128]));
  float e0 = __expf(A[t] - mx), e1 = __expf(A[t + 128] - mx);
  float den = breduce_sum(red, t, e0 + e1);
  float inv = 1.0f / den;
  A[t] = e0 * inv;
  A[t + 128] = e1 * inv;
  __syncthreads();
  // X_int[b,:] += sum_j A[j] * tanh(daI + pb[j])   (also sums over i via atomics)
  float acc = 0.f;
  float da = daI[t];
  for (int j = 0; j < 256; j++)
    acc += A[j] * tanhf(da + pbM[(b * 256 + j) * 128 + t]);
  atomicAdd(&xint[b * 128 + t], acc);
}

// -------------------- pooling / small tails --------------------------------
__global__ void mean_pool(const float* __restrict__ src, float* __restrict__ out,
                          int rows) {
  int b = blockIdx.x, t = threadIdx.x;
  float s = 0.f;
  for (int r = 0; r < rows; r++) s += src[(b * rows + r) * 128 + t];
  out[b * 128 + t] = s / (float)rows;
}

__global__ void zero_kernel(float* __restrict__ p, int n) {
  int i = blockIdx.x * blockDim.x + threadIdx.x;
  if (i < n) p[i] = 0.f;
}

// InfoNCE contrastive loss (B=2 -> 4x4 sim), one block of 128 threads
__global__ void contrast_kernel(const float* __restrict__ h1dpool,
                                const float* __restrict__ h3dpool,
                                const float* __restrict__ w11, const float* __restrict__ b11,
                                const float* __restrict__ w12, const float* __restrict__ b12,
                                const float* __restrict__ w31, const float* __restrict__ b31,
                                const float* __restrict__ w32, const float* __restrict__ b32,
                                float* __restrict__ out_loss) {
  __shared__ float buf[128], hid[128], z[4][128], red[128], simL[16];
  int t = threadIdx.x;
  for (int b = 0; b < 2; b++) {
    buf[t] = h1dpool[b * 128 + t]; __syncthreads();
    float a = 0.f;
    for (int k = 0; k < 128; k++) a += buf[k] * w11[k * 128 + t];
    hid[t] = fmaxf(a + b11[t], 0.f); __syncthreads();
    float zz = 0.f;
    for (int k = 0; k < 128; k++) zz += hid[k] * w12[k * 128 + t];
    z[b][t] = zz + b12[t]; __syncthreads();
    buf[t] = h3dpool[b * 128 + t]; __syncthreads();
    a = 0.f;
    for (int k = 0; k < 128; k++) a += buf[k] * w31[k * 128 + t];
    hid[t] = fmaxf(a + b31[t], 0.f); __syncthreads();
    zz = 0.f;
    for (int k = 0; k < 128; k++) zz += hid[k] * w32[k * 128 + t];
    z[2 + b][t] = zz + b32[t]; __syncthreads();
  }
  for (int r = 0; r < 4; r++) {
    float n2 = breduce_sum(red, t, z[r][t] * z[r][t]);
    z[r][t] *= rsqrtf(n2);
    __syncthreads();
  }
  if (t < 16) {
    int r = t >> 2, c = t & 3;
    float s = 0.f;
    for (int k = 0; k < 128; k++) s += z[r][k] * z[c][k];
    s *= (1.0f / 0.07f);                 // /TEMP
    if (r == c) s = -9.0e15f;            // NEG_DIAG
    simL[t] = s;
  }
  __syncthreads();
  if (t == 0) {
    const int lab[4] = {2, 3, 0, 1};     // labels = [B..2B-1, 0..B-1]
    float loss = 0.f;
    for (int r = 0; r < 4; r++) {
      float m = simL[r * 4];
      for (int c = 1; c < 4; c++) m = fmaxf(m, simL[r * 4 + c]);
      float se = 0.f;
      for (int c = 0; c < 4; c++) se += __expf(simL[r * 4 + c] - m);
      float lse = __logf(se) + m;
      loss -= (simL[r * 4 + lab[r]] - lse);
    }
    out_loss[0] = loss * 0.25f;
  }
}

// predictor MLP, one block of 256 threads, B=2
__global__ void pred_kernel(const float* __restrict__ xint,
                            const float* __restrict__ dpool,
                            const float* __restrict__ hpool,
                            const float* __restrict__ w1, const float* __restrict__ b1,
                            const float* __restrict__ w2, const float* __restrict__ b2,
                            const float* __restrict__ w3, const float* __restrict__ b3,
                            const float* __restrict__ w4, const float* __restrict__ b4,
                            float* __restrict__ out) {
  __shared__ float comb[384], y1[256], y2[128], y3[64];
  int t = threadIdx.x;
  for (int b = 0; b < 2; b++) {
    if (t < 128) {
      comb[t]       = xint[b * 128 + t];
      comb[128 + t] = dpool[b * 128 + t];
      comb[256 + t] = hpool[b * 128 + t];
    }
    __syncthreads();
    {
      float a = 0.f;
      for (int k = 0; k < 384; k++) a += comb[k] * w1[k * 256 + t];
      y1[t] = fmaxf(a + b1[t], 0.f);
    }
    __syncthreads();
    if (t < 128) {
      float a = 0.f;
      for (int k = 0; k < 256; k++) a += y1[k] * w2[k * 128 + t];
      y2[t] = fmaxf(a + b2[t], 0.f);
    }
    __syncthreads();
    if (t < 64) {
      float a = 0.f;
      for (int k = 0; k < 128; k++) a += y2[k] * w3[k * 64 + t];
      y3[t] = fmaxf(a + b3[t], 0.f);
    }
    __syncthreads();
    if (t == 0) {
      float a = 0.f;
      for (int k = 0; k < 64; k++) a += y3[k] * w4[k];
      out[b] = a + b4[0];
    }
    __syncthreads();
  }
}

// ---------------------------------------------------------------------------
extern "C" void kernel_launch(void* const* d_in, const int* in_sizes, int n_in,
                              void* d_out, int out_size, void* d_ws, size_t ws_size,
                              hipStream_t stream) {
  (void)in_sizes; (void)n_in; (void)out_size; (void)ws_size;
  const float* h1d  = (const float*)d_in[0];
  const float* drug = (const float*)d_in[1];
  const float* nf   = (const float*)d_in[2];
  const float* adj  = (const float*)d_in[3];
  const float* dist = (const float*)d_in[4];
  auto P = [&](int i) { return (const float*)d_in[i]; };

  float* wsp = (float*)d_ws;
  size_t off = 0;
  auto alloc = [&](size_t n) { float* p = wsp + off; off += n; return p; };
  float* hc    = alloc(512 * 512);   // dense-concat GNN buffer (B*N, 4H)
  float* a_i   = alloc(512 * 128);
  float* a_jB  = alloc(512 * 128);
  float* si    = alloc(512);
  float* sj    = alloc(512);
  float* pagg  = alloc(512 * 128);
  float* sumw  = alloc(512);
  float* aggb  = alloc(512 * 128);
  float* tmp1  = alloc(512 * 128);
  float* h3d   = alloc(512 * 128);
  float* qb    = alloc(512 * 128);
  float* kb    = alloc(1024 * 128);
  float* vb    = alloc(1024 * 128);
  float* ob    = alloc(512 * 128);
  float* xb    = alloc(512 * 128);
  float* hps   = alloc(512 * 128);
  float* dpr   = alloc(128 * 128);
  float* dar   = alloc(128 * 128);
  float* ppr   = alloc(512 * 128);
  float* pbr   = alloc(512 * 128);
  float* xint  = alloc(256);
  float* pl1   = alloc(256);
  float* pl3   = alloc(256);
  float* plh   = alloc(256);
  float* pld   = alloc(256);

  const dim3 g512(8, 2), g1024(16, 2), g128(2, 2);

  // ---- GNN ----
  gnn_in_kernel<<<512, 128, 0, stream>>>(nf, P(24), P(23), hc);
  for (int l = 0; l < 3; l++) {
    int in_d = 128 * (l + 1);
    int base = 25 + 8 * l;
    const float* W1 = P(base + 0); const float* W2 = P(base + 1);
    const float* Wa = P(base + 2); const float* Wu = P(base + 3);
    const float* b1 = P(base + 4); const float* b2 = P(base + 5);
    const float* ba = P(base + 6); const float* bu = P(base + 7);
    gemm_wmma<<<g512, 128, 0, stream>>>(hc, 512, W1, 128, nullptr, nullptr, 0,
                                        a_i, 128, in_d, 0);
    gemm_wmma<<<g512, 128, 0, stream>>>(hc, 512, W1 + (size_t)in_d * 128, 128,
                                        nullptr, nullptr, 0, a_jB, 128, in_d, 0);
    attn_proj_kernel<<<512, 128, 0, stream>>>(hc, Wa, in_d, si, sj);
    gnn_edge_kernel<<<512, 128, 0, stream>>>(si, sj, dist, adj, a_i, a_jB,
                                             W1 + (size_t)2 * in_d * 128, b1,
                                             Wa + 2 * in_d, ba, pagg, sumw);
    gemm_wmma<<<g512, 128, 0, stream>>>(pagg, 128, W2, 128, nullptr, nullptr, 0,
                                        aggb, 128, 128, 0);
    addsb_kernel<<<512, 128, 0, stream>>>(aggb, sumw, b2);
    gemm_wmma<<<g512, 128, 0, stream>>>(aggb, 128, Wu + (size_t)in_d * 128, 128,
                                        bu, nullptr, 0, tmp1, 128, 128, 0);
    gemm_wmma<<<g512, 128, 0, stream>>>(hc, 512, Wu, 128, nullptr, tmp1, 128,
                                        hc + 128 * (l + 1), 512, in_d, 1);
  }
  gemm_wmma<<<g512, 128, 0, stream>>>(hc, 512, P(50), 128, P(49), nullptr, 0,
                                      h3d, 128, 512, 0);

  // ---- cross attention (Q=3D, K=V=1D) + residual LN ----
  gemm_wmma<<<g512, 128, 0, stream>>>(h3d, 128, P(21), 128, P(15), nullptr, 0,
                                      qb, 128, 128, 0);
  gemm_wmma<<<g1024, 128, 0, stream>>>(h1d, 128, P(19), 128, P(13), nullptr, 0,
                                       kb, 128, 128, 0);
  gemm_wmma<<<g1024, 128, 0, stream>>>(h1d, 128, P(22), 128, P(16), nullptr, 0,
                                       vb, 128, 128, 0);
  mha_kernel<<<4096, 128, 0, stream>>>(qb, kb, vb, ob);
  gemm_wmma<<<g512, 128, 0, stream>>>(ob, 128, P(20), 128, P(14), nullptr, 0,
                                      xb, 128, 128, 0);
  ln_kernel<<<512, 128, 0, stream>>>(xb, h3d, P(18), P(17), hps);

  // ---- joint attention ----
  gemm_wmma<<<g128, 128, 0, stream>>>(drug, 128, P(53), 128, nullptr, nullptr, 0,
                                      dpr, 128, 128, 0);
  gemm_wmma<<<g128, 128, 0, stream>>>(drug, 128, P(51), 128, nullptr, nullptr, 0,
                                      dar, 128, 128, 0);
  gemm_wmma<<<g512, 128, 0, stream>>>(hps, 128, P(54), 128, nullptr, nullptr, 0,
                                      ppr, 128, 128, 0);
  gemm_wmma<<<g512, 128, 0, stream>>>(hps, 128, P(52), 128, nullptr, nullptr, 0,
                                      pbr, 128, 128, 0);
  zero_kernel<<<1, 256, 0, stream>>>(xint, 256);
  joint_kernel<<<128, 128, 0, stream>>>(dpr, ppr, dar, pbr, P(55), xint);

  // ---- pools ----
  mean_pool<<<2, 128, 0, stream>>>(h1d, pl1, 512);
  mean_pool<<<2, 128, 0, stream>>>(h3d, pl3, 256);
  mean_pool<<<2, 128, 0, stream>>>(hps, plh, 256);
  mean_pool<<<2, 128, 0, stream>>>(drug, pld, 64);

  // ---- heads ----
  float* out = (float*)d_out;
  contrast_kernel<<<1, 128, 0, stream>>>(pl1, pl3, P(9), P(5), P(10), P(6),
                                         P(11), P(7), P(12), P(8), out + 2);
  pred_kernel<<<1, 256, 0, stream>>>(xint, pld, plh, P(60), P(56), P(61), P(57),
                                     P(62), P(58), P(63), P(59), out);
}
